// GATLayer_73555609911760
// MI455X (gfx1250) — compile-verified, matrix operand order checked by reading, not run on hardware
//
#include <hip/hip_runtime.h>
#include <hip/hip_bf16.h>

// ---------------------------------------------------------------------------
// GAT layer forward, fused for MI455X (gfx1250, wave32, WMMA).
//   out0  = relu( (softmax_m(mask(e)) @ Wh) transposed )   (B,N,H*Dout)
//   alpha = softmax over m of e, e = f1[n]+f2[m] masked    (B,H,N,N)
// f1 is constant along the softmax axis -> cancels; never computed.
// exp(f2-mx[r]) = exp(f2-gmax)*exp(gmax-mx[r]) -> one exp table per (b,h)
// block (stable via block max), zero exps in the streaming phases.
// adj read once (NT loads) + alpha written once (NT stores): the two 537MB
// streams bypass L2 retention so the 8MB WhB fragment buffer stays resident.
// HBM floor: ~1.1GB ~ 46us @ 23.3 TB/s.
// ---------------------------------------------------------------------------

typedef _Float16     h4   __attribute__((ext_vector_type(4)));
typedef _Float16     h8   __attribute__((ext_vector_type(8)));
typedef _Float16     v16h __attribute__((ext_vector_type(16)));
typedef float        v8f  __attribute__((ext_vector_type(8)));
typedef int          i4   __attribute__((ext_vector_type(4)));
typedef unsigned int u4   __attribute__((ext_vector_type(4)));
typedef float        f4   __attribute__((ext_vector_type(4)));

// ds_swizzle group-of-32 XOR pattern (offset[15]=0, and=0x1f, or=0, xor=m)
#define SWZ_XOR_F(v, m) \
    __int_as_float(__builtin_amdgcn_ds_swizzle(__float_as_int(v), 0x7C00 | (m)))

// 16-bit WMMA fragment k-mapping (ISA 7.12.2): lane group g = lane>>4,
// per-lane half i -> k = (i<8) ? 8*g+i : 8+8*g+i
static __device__ __forceinline__ int frag_k(int g, int i) {
    return (i < 8) ? (8 * g + i) : (8 + 8 * g + i);
}

static __device__ __forceinline__ v16h frag_combine(h8 lo, h8 hi) {
    v16h r;
#pragma unroll
    for (int i = 0; i < 8; ++i) { r[i] = lo[i]; r[i + 8] = hi[i]; }
    return r;
}

static __device__ __forceinline__ v8f wmma_f16(v16h A, v16h B, v8f C) {
    return __builtin_amdgcn_wmma_f32_16x16x32_f16(
        false, A, false, B, (short)0, C, false, false);
}

// ---------------------------------------------------------------------------
// K0: stage W (Hh,256,64) f32 -> f16 WMMA-B fragments.
// Layout: [h][kc(8)][nc(4)][lane(32)][half(16)] contiguous.
// ---------------------------------------------------------------------------
__global__ __launch_bounds__(256)
void gat_stage_w(const float* __restrict__ W, _Float16* __restrict__ Wstage) {
    int t = blockIdx.x * 256 + threadIdx.x;       // 131072 total
    int i    = t & 15;
    int lane = (t >> 4) & 31;
    int nc   = (t >> 9) & 3;
    int kc   = (t >> 11) & 7;
    int h    = t >> 14;
    int g = lane >> 4, e = lane & 15;
    int d   = kc * 32 + frag_k(g, i);             // K index into Din
    int col = nc * 16 + e;                        // output feature
    Wstage[t] = (_Float16)W[(h * 256 + d) * 64 + col];
}

// ---------------------------------------------------------------------------
// K1: Wh[b,h] = H[b] @ W[h] via WMMA; emit Wh as f16 B-fragments for stage 2
// plus f2 = Wh @ a2. One wave = one (b,h, 16-row tile). 4 waves / block.
// Staged Wh layout: [bh(32)][mc(64)][nc(4)][lane(32)][half(16)].
// ---------------------------------------------------------------------------
__global__ __launch_bounds__(128)
void gat_wh(const float* __restrict__ H, const float* __restrict__ a,
            const _Float16* __restrict__ Wstage,
            _Float16* __restrict__ WhB, float* __restrict__ f2ws) {
    __shared__ __align__(16) _Float16 Ht[4][16 * 40];   // padded 16x32 f16 tile

    int lane = threadIdx.x & 31;
    int w    = threadIdx.x >> 5;
    int idx  = blockIdx.x * 4 + w;                 // 0..4095
    int bh   = idx >> 7;                           // 0..31
    int tile = idx & 127;                          // 16-row tile of N
    int b = bh >> 3, h = bh & 7;
    int n0 = tile * 16;
    int g = lane >> 4, e = lane & 15;
    int q = lane & 7, rh = lane >> 3;              // quad / row-subindex

    const float* Hrow = H + ((size_t)b * 2048 + n0) * 256;
    _Float16* ht = Ht[w];
    const h8* wst = (const h8*)Wstage;

    v8f acc[4];
#pragma unroll
    for (int nc = 0; nc < 4; ++nc)
#pragma unroll
        for (int j = 0; j < 8; ++j) acc[nc][j] = 0.0f;

    for (int kc = 0; kc < 8; ++kc) {
        // issue B-fragment loads early (overlap with H staging)
        h8 bf0[4], bf1[4];
        size_t fi0 = (((size_t)h * 8 + kc) * 4) * 64 + lane * 2;
#pragma unroll
        for (int nc = 0; nc < 4; ++nc) {
            bf0[nc] = wst[fi0 + (size_t)nc * 64];
            bf1[nc] = wst[fi0 + (size_t)nc * 64 + 1];
        }

        // stage 16x32 f32 slice of H -> f16 LDS: 4 rows/pass via b128 loads
#pragma unroll
        for (int it = 0; it < 4; ++it) {
            int r = it * 4 + rh;
            f4 hv = *(const f4*)&Hrow[(size_t)r * 256 + kc * 32 + q * 4];
            h4 hc;
            hc[0] = (_Float16)hv.x; hc[1] = (_Float16)hv.y;
            hc[2] = (_Float16)hv.z; hc[3] = (_Float16)hv.w;
            *(h4*)&ht[r * 40 + q * 4] = hc;        // ds_store_b64
        }
        asm volatile("" ::: "memory");   // LDS is in-order within a wave

        // A fragment: row = lane&15, halves via frag_k -> 2x ds_load_b128
        h8 lo = *(const h8*)&ht[(lane & 15) * 40 + 8 * g];
        h8 hi = *(const h8*)&ht[(lane & 15) * 40 + 16 + 8 * g];
        v16h A = frag_combine(lo, hi);

#pragma unroll
        for (int nc = 0; nc < 4; ++nc)
            acc[nc] = wmma_f16(A, frag_combine(bf0[nc], bf1[nc]), acc[nc]);
    }

    // Epilogue: C-frag (lane=col e, vgpr v=row 8g+v) -> staged B halves p*8+v
    int mc = tile >> 1, p = tile & 1;
    h8* whb = (h8*)WhB;
    float part[8];
#pragma unroll
    for (int v = 0; v < 8; ++v) part[v] = 0.0f;

#pragma unroll
    for (int nc = 0; nc < 4; ++nc) {
        float a2 = a[h * 128 + 64 + nc * 16 + e];
        h8 hv;
#pragma unroll
        for (int v = 0; v < 8; ++v) {
            float c = acc[nc][v];
            hv[v] = (_Float16)c;
            part[v] += c * a2;
        }
        size_t oi = ((((size_t)bh * 64 + mc) * 4 + nc) * 64) + lane * 2 + p;
        whb[oi] = hv;                               // global_store_b128
    }

    // f2[n0+8g+v] = sum over e-lanes (xor 1,2,4,8 stays inside 16-lane group)
#pragma unroll
    for (int v = 0; v < 8; ++v) {
        float s = part[v];
        s += SWZ_XOR_F(s, 1);
        s += SWZ_XOR_F(s, 2);
        s += SWZ_XOR_F(s, 4);
        s += SWZ_XOR_F(s, 8);
        if (e == 0) f2ws[(size_t)bh * 2048 + n0 + 8 * g + v] = s;
    }
}

// ---------------------------------------------------------------------------
// K2: fused mask/softmax/alpha-write + alpha@Wh + relu.
// Block = 4 waves = 64 rows of one (b,h); wave = 16 rows.
// adj read exactly once (NT int4, mask cached as ballot bits in LDS);
// alpha written once (NT b128); zero exps in the streaming loops.
// ---------------------------------------------------------------------------
__global__ __launch_bounds__(128)
void gat_main(const int* __restrict__ adj, const _Float16* __restrict__ WhB,
              const float* __restrict__ f2ws, float* __restrict__ out0,
              float* __restrict__ alpha_out) {
    __shared__ __align__(16) float        f2s[2048];          // 8KB
    __shared__ __align__(16) float        expf2s[2048];       // 8KB exp table
    __shared__ __align__(16) unsigned int maskb[4][16][64];   // 16KB mask bits
    __shared__ __align__(16) _Float16     atile[4][16 * 40];  // alpha f16 tile
    __shared__ float2 sc[4][16];                              // {1/sum, unif}
    __shared__ float  gred[4];

    int lane = threadIdx.x & 31;
    int w    = threadIdx.x >> 5;
    int bh   = blockIdx.x >> 5;
    int t64  = blockIdx.x & 31;
    int b = bh >> 3, h = bh & 7;
    int n0 = t64 * 64 + w * 16;
    int g = lane >> 4, e = lane & 15;
    int q = lane & 7, rh = lane >> 3;              // quad / row-subindex

    // cooperative load of f2 row (2048 f32) into LDS
    {
        const f4* src = (const f4*)(f2ws + (size_t)bh * 2048);
        f4* dst = (f4*)f2s;
        for (int i = threadIdx.x; i < 512; i += 128) dst[i] = src[i];
    }
    __syncthreads();

    // block max of f2 (>= any masked row max -> stable exp table)
    {
        float gm = -3.0e38f;
        for (int i = threadIdx.x; i < 2048; i += 128) gm = fmaxf(gm, f2s[i]);
        gm = fmaxf(gm, SWZ_XOR_F(gm, 1));
        gm = fmaxf(gm, SWZ_XOR_F(gm, 2));
        gm = fmaxf(gm, SWZ_XOR_F(gm, 4));
        gm = fmaxf(gm, SWZ_XOR_F(gm, 8));
        gm = fmaxf(gm, SWZ_XOR_F(gm, 16));
        if (lane == 0) gred[w] = gm;
        __syncthreads();
        gm = fmaxf(fmaxf(gred[0], gred[1]), fmaxf(gred[2], gred[3]));
        for (int i = threadIdx.x; i < 2048; i += 128)
            expf2s[i] = __expf(f2s[i] - gm);       // only 2048 exps per block
        __syncthreads();
    }

    const int* adjb = adj + (size_t)b * 2048 * 2048;

    // ---- Phase A: stream adj once: ballot-cache mask + masked sum of exp
    for (int r = 0; r < 16; ++r) {
        const i4* arow4 = (const i4*)(adjb + (size_t)(n0 + r) * 2048);
        if (b * 2048 + n0 + r < 4 * 2048 - 1)      // prefetch next adj row
            __builtin_prefetch((const int*)arow4 + 2048 + lane * 64, 0, 0);

        float s = 0.0f;
        unsigned int any = 0u;
        for (int it = 0; it < 16; ++it) {
            i4 av = __builtin_nontemporal_load(&arow4[it * 32 + lane]); // NT
            f4 ef = *(const f4*)&expf2s[it * 128 + lane * 4];
            unsigned int m0 = __builtin_amdgcn_ballot_w32(av.x != 0);
            unsigned int m1 = __builtin_amdgcn_ballot_w32(av.y != 0);
            unsigned int m2 = __builtin_amdgcn_ballot_w32(av.z != 0);
            unsigned int m3 = __builtin_amdgcn_ballot_w32(av.w != 0);
            any |= (m0 | m1 | m2 | m3);
            if (lane == 0) {
                u4 m4;
                m4.x = m0; m4.y = m1; m4.z = m2; m4.w = m3;
                *(u4*)&maskb[w][r][it * 4] = m4;
            }
            s += (av.x != 0 ? ef.x : 0.0f);
            s += (av.y != 0 ? ef.y : 0.0f);
            s += (av.z != 0 ? ef.z : 0.0f);
            s += (av.w != 0 ? ef.w : 0.0f);
        }
        s += SWZ_XOR_F(s, 1);
        s += SWZ_XOR_F(s, 2);
        s += SWZ_XOR_F(s, 4);
        s += SWZ_XOR_F(s, 8);
        s += SWZ_XOR_F(s, 16);
        if (lane == 0) {
            bool hasany = (any != 0u);
            float2 v;
            v.x = hasany ? 1.0f / s : 0.0f;                // row scale
            v.y = hasany ? 0.0f : (1.0f / 2048.0f);        // isolated-node row
            sc[w][r] = v;
        }
    }
    asm volatile("" ::: "memory");

    // ---- Phase B: alpha write (NT b128) + alpha(f16) @ Wh(f16) via WMMA
    _Float16* at = atile[w];
    float* aout = alpha_out + (((size_t)bh) * 2048 + n0) * 2048;
    const h8* whb = (const h8*)WhB;

    v8f acc[4];
#pragma unroll
    for (int nc = 0; nc < 4; ++nc)
#pragma unroll
        for (int j = 0; j < 8; ++j) acc[nc][j] = 0.0f;

    for (int kc = 0; kc < 64; ++kc) {
        // issue B-fragment loads early; latency hides behind alpha generation
        h8 bf0[4], bf1[4];
        size_t fb = (((size_t)bh * 64 + kc) * 4) * 64 + lane * 2;
#pragma unroll
        for (int nc = 0; nc < 4; ++nc) {
            bf0[nc] = whb[fb + (size_t)nc * 64];
            bf1[nc] = whb[fb + (size_t)nc * 64 + 1];
        }

        f4 ev = *(const f4*)&expf2s[kc * 32 + q * 4];
        int mdw = (kc >> 2) * 4;                   // mask dword group
        int sh  = (kc & 3) * 8 + q;                // mask bit index
#pragma unroll
        for (int it = 0; it < 4; ++it) {           // 4 rows per pass
            int r = it * 4 + rh;
            u4     md  = *(const u4*)&maskb[w][r][mdw];     // LDS broadcast
            float2 scv = sc[w][r];
            f4 al;
            al.x = (((md.x >> sh) & 1u) ? ev.x * scv.x : 0.0f) + scv.y;
            al.y = (((md.y >> sh) & 1u) ? ev.y * scv.x : 0.0f) + scv.y;
            al.z = (((md.z >> sh) & 1u) ? ev.z * scv.x : 0.0f) + scv.y;
            al.w = (((md.w >> sh) & 1u) ? ev.w * scv.x : 0.0f) + scv.y;
            __builtin_nontemporal_store(
                al, (f4*)&aout[(size_t)r * 2048 + kc * 32 + q * 4]); // NT
            h4 hv;
            hv[0] = (_Float16)al.x; hv[1] = (_Float16)al.y;
            hv[2] = (_Float16)al.z; hv[3] = (_Float16)al.w;
            *(h4*)&at[r * 40 + q * 4] = hv;        // ds_store_b64
        }
        asm volatile("" ::: "memory");

        h8 lo = *(const h8*)&at[e * 40 + 8 * g];   // A frag: row = e
        h8 hi = *(const h8*)&at[e * 40 + 16 + 8 * g];
        v16h A = frag_combine(lo, hi);

#pragma unroll
        for (int nc = 0; nc < 4; ++nc)
            acc[nc] = wmma_f16(A, frag_combine(bf0[nc], bf1[nc]), acc[nc]);
    }

    // ---- Epilogue: relu + scatter into (B,N,H*Dout)
    float* orow = out0 + ((size_t)b * 2048 + n0) * 512 + (size_t)h * 64;
#pragma unroll
    for (int nc = 0; nc < 4; ++nc)
#pragma unroll
        for (int v = 0; v < 8; ++v)
            orow[(size_t)(8 * g + v) * 512 + nc * 16 + e] =
                fmaxf(acc[nc][v], 0.0f);
}

// ---------------------------------------------------------------------------
extern "C" void kernel_launch(void* const* d_in, const int* in_sizes, int n_in,
                              void* d_out, int out_size, void* d_ws, size_t ws_size,
                              hipStream_t stream) {
    (void)in_sizes; (void)n_in; (void)out_size; (void)ws_size;
    const float* H   = (const float*)d_in[0];   // (4,2048,256)
    const int*   adj = (const int*)d_in[1];     // (4,2048,2048)
    const float* W   = (const float*)d_in[2];   // (8,256,64)
    const float* a   = (const float*)d_in[3];   // (8,128)

    float* out0  = (float*)d_out;                        // (4,2048,512)
    float* alpha = out0 + (size_t)4 * 2048 * 512;        // (4,8,2048,2048)

    // workspace: Wstage 256KB | WhB 8MB | f2 256KB  (= 8.9MB total)
    char* ws = (char*)d_ws;
    _Float16* wstage = (_Float16*)ws;
    _Float16* whb    = (_Float16*)(ws + 262144);
    float*    f2ws   = (float*)(ws + 262144 + 8388608);

    gat_stage_w<<<512, 256, 0, stream>>>(W, wstage);
    gat_wh<<<1024, 128, 0, stream>>>(H, a, wstage, whb, f2ws);
    gat_main<<<1024, 128, 0, stream>>>(adj, whb, f2ws, out0, alpha);
}
